// Attention_40407052321024
// MI455X (gfx1250) — matrix operand order repaired and checked by reading, not compiled, not run on hardware
//
#include <hip/hip_runtime.h>
#include <hip/hip_bf16.h>

// ---------------------------------------------------------------------------
// Problem constants (from the reference)
// ---------------------------------------------------------------------------
#define BB   16
#define TDEC 1024
#define TTXT 512
#define DM   1024
#define SPKN 256

typedef __bf16 bf16_t;
typedef bf16_t v16bf __attribute__((ext_vector_type(16)));
typedef float  v8f   __attribute__((ext_vector_type(8)));

struct alignas(16) H8 { unsigned short h[8]; };   // 8 bf16 = 16 bytes
union FragU { H8 half8[2]; v16bf v; };            // 16 bf16 = one WMMA operand

__device__ __forceinline__ unsigned short f2bf(float f) {
  union { float f; unsigned u; } c; c.f = f;
  unsigned u = c.u;
  u += 0x7FFFu + ((u >> 16) & 1u);                // round-to-nearest-even
  return (unsigned short)(u >> 16);
}

__device__ __forceinline__ void stv(float* p, float v)          { *p = v; }
__device__ __forceinline__ void stv(unsigned short* p, float v) { *p = f2bf(v); }

// ---------------------------------------------------------------------------
// 1) speaker rates: rate = sigmoid(spk . W + b), per batch
// ---------------------------------------------------------------------------
__global__ void rates_kernel(const float* __restrict__ spk,
                             const float* __restrict__ Wsq, const float* __restrict__ bsq,
                             const float* __restrict__ Wsk, const float* __restrict__ bsk,
                             float* __restrict__ rate_q, float* __restrict__ rate_k) {
  __shared__ float sq[256];
  __shared__ float sk[256];
  const int b = blockIdx.x, t = threadIdx.x;
  const float se = spk[b * SPKN + t];
  sq[t] = se * Wsq[t];
  sk[t] = se * Wsk[t];
  __syncthreads();
  for (int s = 128; s > 0; s >>= 1) {
    if (t < s) { sq[t] += sq[t + s]; sk[t] += sk[t + s]; }
    __syncthreads();
  }
  if (t == 0) {
    rate_q[b] = 1.f / (1.f + __expf(-(sq[0] + bsq[0])));
    rate_k[b] = 1.f / (1.f + __expf(-(sk[0] + bsk[0])));
  }
}

// ---------------------------------------------------------------------------
// 2) fp32 -> bf16 convert (weights, values)
// ---------------------------------------------------------------------------
__global__ void cvt_f32_bf16(const float* __restrict__ x, unsigned short* __restrict__ y, long n) {
  long i = (long)blockIdx.x * blockDim.x + threadIdx.x;
  if (i < n) y[i] = f2bf(x[i]);
}

// ---------------------------------------------------------------------------
// 3) x + rate-scaled sinusoidal PE, fused bf16 convert.
//    even feature -> sin, odd -> cos; angle = rate[b]*pos / 10000^(k/D)
// ---------------------------------------------------------------------------
__global__ void posenc_kernel(const float* __restrict__ x, const float* __restrict__ rate,
                              const int* __restrict__ pos0, unsigned short* __restrict__ y,
                              int T, long n) {
  long i = (long)blockIdx.x * blockDim.x + threadIdx.x;
  if (i >= n) return;
  const int  kd = (int)(i % DM);
  const long bt = i / DM;
  const int  t  = (int)(bt % T);
  const int  b  = (int)(bt / T);
  const float pos    = (float)(t + (pos0 ? pos0[0] : 0));
  const float invdiv = __expf(-((float)kd / (float)DM) * 9.210340371976184f); // ln(10000)
  const float ang    = rate[b] * pos * invdiv;
  const float pe     = (kd & 1) ? __cosf(ang) : __sinf(ang);
  y[i] = f2bf(x[i] + pe);
}

// ---------------------------------------------------------------------------
// 4) bf16 WMMA GEMM:  Y[b] = scale * (X[b] (MxK) . Bmat[b]) + bias
//    Bmat element (k,n) at Bm + k*ldbK + n*ldbN  (BKCONTIG => ldbK==1,
//    which covers X.W^T / q.k^T; BKCONTIG=false covers P.V).
//    Block tile 128x64, BK=32, 256 threads = 8 waves, each wave owns a
//    2x2 grid of 16x16 WMMA accumulators. B staged in LDS pre-swizzled so a
//    lane's fragment is two contiguous ds_load_b128 in contiguous k order (k = 16*(l>>4) + s):
//      A (16x32 bf16): lane l -> m=l&15, koff=(l>>4)*16, K = koff..+15
//      B (32x16 bf16): lane l -> n=l&15, VGPR r holds K = (l>>4)*16 + 2r,2r+1
//      C (16x16 f32) : VGPR r -> M = r + 8*(l>>4), N = l&15
// ---------------------------------------------------------------------------
template <typename OutT, bool BKCONTIG>
__global__ void __launch_bounds__(256)
gemm_bf16_wmma(const unsigned short* __restrict__ X, long xBatch, int lda,
               const unsigned short* __restrict__ Bm, long bBatch, int ldbK, int ldbN,
               OutT* __restrict__ Y, long yBatch, int ldy,
               const float* __restrict__ bias, float scale, int Ksize) {
  constexpr int BM = 128, BN = 64, BK = 32;
  __shared__ unsigned short sA[BM * BK];   // row-major [m][k]
  __shared__ unsigned short sB[BN * BK];   // swizzled: ntile*512 + khalf*256 + n_in*16 + kk

  const int tid  = threadIdx.x;
  const int lane = tid & 31;
  const int wv   = tid >> 5;
  const int mw   = (wv & 3) * 32;          // wave's M offset in tile
  const int nw   = (wv >> 2) * 32;         // wave's N offset in tile
  const int batch = blockIdx.z;

  const unsigned short* Xp = X + (long)batch * xBatch + (long)(blockIdx.x * BM) * lda;
  const unsigned short* Bp = Bm + (long)batch * bBatch + (long)(blockIdx.y * BN) * ldbN;

  v8f acc[2][2] = {};

  // cooperative-fill indexing
  const int am  = tid >> 1,  ak0 = (tid & 1) * 16;   // A: 2 threads/row, 16 bf16 each
  const int bn  = tid >> 2,  bk0 = (tid & 3) * 8;    // B: 4 threads/col, 8 bf16 each
  const int bLds = (bn >> 4) * 512 + (bk0 >> 4) * 256 + (bn & 15) * 16 + (bk0 & 15);
  const int koff = (lane >> 4) * 16;

  const int nSteps = Ksize / BK;
  for (int kt = 0; kt < nSteps; ++kt) {
    __syncthreads();
    {   // stage A tile (two b128 per thread, contiguous in K)
      const H8* g = (const H8*)(Xp + (long)am * lda + kt * BK + ak0);
      *(H8*)&sA[am * BK + ak0]     = g[0];
      *(H8*)&sA[am * BK + ak0 + 8] = g[1];
      if (kt + 1 < nSteps)
        __builtin_prefetch(Xp + (long)am * lda + (kt + 1) * BK + ak0, 0, 1);
    }
    {   // stage B tile into swizzled layout
      if constexpr (BKCONTIG) {
        *(H8*)&sB[bLds] = *(const H8*)(Bp + (long)bn * ldbN + kt * BK + bk0);
      } else {
        H8 tmp;
#pragma unroll
        for (int i = 0; i < 8; ++i)
          tmp.h[i] = Bp[(long)(kt * BK + bk0 + i) * ldbK + (long)bn * ldbN];
        *(H8*)&sB[bLds] = tmp;
      }
    }
    __syncthreads();

    FragU a[2], bfr[2];
#pragma unroll
    for (int mi = 0; mi < 2; ++mi) {
      const int row = mw + mi * 16 + (lane & 15);
      a[mi].half8[0] = *(const H8*)&sA[row * BK + koff];
      a[mi].half8[1] = *(const H8*)&sA[row * BK + koff + 8];
    }
#pragma unroll
    for (int ni = 0; ni < 2; ++ni) {
      const int base = ((nw >> 4) + ni) * 512 + (lane >> 4) * 256 + (lane & 15) * 16;
      bfr[ni].half8[0] = *(const H8*)&sB[base];
      bfr[ni].half8[1] = *(const H8*)&sB[base + 8];
    }
#pragma unroll
    for (int mi = 0; mi < 2; ++mi)
#pragma unroll
      for (int ni = 0; ni < 2; ++ni)
        acc[mi][ni] = __builtin_amdgcn_wmma_f32_16x16x32_bf16(
            false, a[mi].v, false, bfr[ni].v, (short)0, acc[mi][ni], false, false);
  }

  // epilogue (documented C layout)
  OutT* Yp = Y + (long)batch * yBatch;
#pragma unroll
  for (int mi = 0; mi < 2; ++mi) {
    const int row0 = blockIdx.x * BM + mw + mi * 16 + (lane >> 4) * 8;
#pragma unroll
    for (int ni = 0; ni < 2; ++ni) {
      const int col = blockIdx.y * BN + nw + ni * 16 + (lane & 15);
      const float bv = bias ? bias[col] : 0.f;
#pragma unroll
      for (int r = 0; r < 8; ++r)
        stv(&Yp[(long)(row0 + r) * ldy + col], acc[mi][ni][r] * scale + bv);
    }
  }
}

// ---------------------------------------------------------------------------
// 5) dense row softmax over 512 fp32 scores -> bf16 probabilities
//    one wave32 per row, 16 elements/lane, shfl_xor tree reductions
// ---------------------------------------------------------------------------
__global__ void __launch_bounds__(256)
softmax_kernel(const float* __restrict__ S, unsigned short* __restrict__ P) {
  const int lane = threadIdx.x & 31;
  const int row  = blockIdx.x * 8 + (threadIdx.x >> 5);
  const float* s = S + (long)row * TTXT;
  float v[16];
  float m = -3.4e38f;
#pragma unroll
  for (int i = 0; i < 16; ++i) { v[i] = s[lane + 32 * i]; m = fmaxf(m, v[i]); }
#pragma unroll
  for (int off = 16; off > 0; off >>= 1) m = fmaxf(m, __shfl_xor(m, off, 32));
  float sum = 0.f;
#pragma unroll
  for (int i = 0; i < 16; ++i) { v[i] = __expf(v[i] - m); sum += v[i]; }
#pragma unroll
  for (int off = 16; off > 0; off >>= 1) sum += __shfl_xor(sum, off, 32);
  const float inv = 1.f / sum;
  unsigned short* p = P + (long)row * TTXT;
#pragma unroll
  for (int i = 0; i < 16; ++i) p[lane + 32 * i] = f2bf(v[i] * inv);
}

// ---------------------------------------------------------------------------
// launch
// ---------------------------------------------------------------------------
extern "C" void kernel_launch(void* const* d_in, const int* in_sizes, int n_in,
                              void* d_out, int out_size, void* d_ws, size_t ws_size,
                              hipStream_t stream) {
  (void)in_sizes; (void)n_in; (void)out_size; (void)ws_size;

  const float* query  = (const float*)d_in[0];
  const float* keys   = (const float*)d_in[1];
  const float* values = (const float*)d_in[2];
  const float* spk    = (const float*)d_in[3];
  const float* Wsq    = (const float*)d_in[4];
  const float* bsq    = (const float*)d_in[5];
  const float* Wsk    = (const float*)d_in[6];
  const float* bsk    = (const float*)d_in[7];
  const float* Wq     = (const float*)d_in[8];
  const float* bq     = (const float*)d_in[9];
  const float* Wk     = (const float*)d_in[10];
  const float* bk     = (const float*)d_in[11];
  const float* Wv     = (const float*)d_in[12];
  const float* bv     = (const float*)d_in[13];
  const float* Wo     = (const float*)d_in[14];
  const float* bo     = (const float*)d_in[15];
  const int*   mpos   = (const int*)d_in[16];
  float*       out    = (float*)d_out;

  // workspace carve-up (256B aligned)
  char* w = (char*)d_ws;
  size_t off = 0;
  auto alloc = [&](size_t bytes) { size_t o = off; off += (bytes + 255) & ~(size_t)255; return o; };

  float*          rate_q = (float*)(w + alloc(BB * sizeof(float)));
  float*          rate_k = (float*)(w + alloc(BB * sizeof(float)));
  unsigned short* qpe  = (unsigned short*)(w + alloc((size_t)BB * TDEC * DM * 2));
  unsigned short* kpe  = (unsigned short*)(w + alloc((size_t)BB * TTXT * DM * 2));
  unsigned short* vbf  = (unsigned short*)(w + alloc((size_t)BB * TTXT * DM * 2));
  unsigned short* wq16 = (unsigned short*)(w + alloc((size_t)DM * DM * 2));
  unsigned short* wk16 = (unsigned short*)(w + alloc((size_t)DM * DM * 2));
  unsigned short* wv16 = (unsigned short*)(w + alloc((size_t)DM * DM * 2));
  unsigned short* wo16 = (unsigned short*)(w + alloc((size_t)DM * DM * 2));
  unsigned short* qp   = (unsigned short*)(w + alloc((size_t)BB * TDEC * DM * 2));
  unsigned short* kp   = (unsigned short*)(w + alloc((size_t)BB * TTXT * DM * 2));
  unsigned short* vp   = (unsigned short*)(w + alloc((size_t)BB * TTXT * DM * 2));
  float*          sco  = (float*)(w + alloc((size_t)BB * TDEC * TTXT * 4));
  unsigned short* pmat = (unsigned short*)(w + alloc((size_t)BB * TDEC * TTXT * 2));
  unsigned short* attn = (unsigned short*)(w + alloc((size_t)BB * TDEC * DM * 2));

  const float invsq = 0.03125f;  // 1/sqrt(1024)

  // 1) speaker rates
  rates_kernel<<<BB, 256, 0, stream>>>(spk, Wsq, bsq, Wsk, bsk, rate_q, rate_k);

  // 2) bf16 converts
  const long wn = (long)DM * DM;
  cvt_f32_bf16<<<(int)(wn / 256), 256, 0, stream>>>(Wq, wq16, wn);
  cvt_f32_bf16<<<(int)(wn / 256), 256, 0, stream>>>(Wk, wk16, wn);
  cvt_f32_bf16<<<(int)(wn / 256), 256, 0, stream>>>(Wv, wv16, wn);
  cvt_f32_bf16<<<(int)(wn / 256), 256, 0, stream>>>(Wo, wo16, wn);
  const long vn = (long)BB * TTXT * DM;
  cvt_f32_bf16<<<(int)(vn / 256), 256, 0, stream>>>(values, vbf, vn);

  // 3) positional encodings (keys at pos 0, query at current_mel_pos)
  const long kn = (long)BB * TTXT * DM;
  const long qn = (long)BB * TDEC * DM;
  posenc_kernel<<<(int)(kn / 256), 256, 0, stream>>>(keys, rate_k, nullptr, kpe, TTXT, kn);
  posenc_kernel<<<(int)(qn / 256), 256, 0, stream>>>(query, rate_q, mpos, qpe, TDEC, qn);

  // 4) projections: x . W^T + b   (flatten batch into M, weights shared)
  gemm_bf16_wmma<unsigned short, true><<<dim3((BB * TDEC) / 128, DM / 64, 1), 256, 0, stream>>>(
      qpe, 0, DM, wq16, 0, 1, DM, qp, 0, DM, bq, 1.f, DM);
  gemm_bf16_wmma<unsigned short, true><<<dim3((BB * TTXT) / 128, DM / 64, 1), 256, 0, stream>>>(
      kpe, 0, DM, wk16, 0, 1, DM, kp, 0, DM, bk, 1.f, DM);
  gemm_bf16_wmma<unsigned short, true><<<dim3((BB * TTXT) / 128, DM / 64, 1), 256, 0, stream>>>(
      vbf, 0, DM, wv16, 0, 1, DM, vp, 0, DM, bv, 1.f, DM);

  // 5) scores[b] = q[b] . k[b]^T / sqrt(d)   (batched over z)
  gemm_bf16_wmma<float, true><<<dim3(TDEC / 128, TTXT / 64, BB), 256, 0, stream>>>(
      qp, (long)TDEC * DM, DM, kp, (long)TTXT * DM, 1, DM,
      sco, (long)TDEC * TTXT, TTXT, nullptr, invsq, DM);

  // 6) dense softmax -> bf16 P
  softmax_kernel<<<(BB * TDEC) / 8, 256, 0, stream>>>(sco, pmat);

  // 7) attn_out[b] = P[b] . V[b] / sqrt(d)   (B operand is K-major => strided path)
  gemm_bf16_wmma<unsigned short, false><<<dim3(TDEC / 128, DM / 64, BB), 256, 0, stream>>>(
      pmat, (long)TDEC * TTXT, TTXT, vp, (long)TTXT * DM, DM, 1,
      attn, (long)TDEC * DM, DM, nullptr, invsq, TTXT);

  // 8) out = attn_out . Wo^T + bo  (fp32 out, straight to d_out)
  gemm_bf16_wmma<float, true><<<dim3((BB * TDEC) / 128, DM / 64, 1), 256, 0, stream>>>(
      attn, 0, DM, wo16, 0, 1, DM, out, 0, DM, bo, 1.f, DM);
}
